// MultiheadAttention_18090402251195
// MI455X (gfx1250) — matrix operand order repaired and checked by reading, not compiled, not run on hardware
//
#include <hip/hip_runtime.h>
#include <hip/hip_bf16.h>

// MI455X / gfx1250 multihead attention: T=2048, B=8, C=512, H=8, D=64.
// GEMMs via v_wmma_f32_16x16x32_f16 (wave32 WMMA); K-tile staging via
// gfx1250 GLOBAL_LOAD_ASYNC_TO_LDS (ASYNCcnt) using inline asm (portable
// across toolchains whose builtin prototypes differ).

typedef __attribute__((ext_vector_type(16))) _Float16 v16h;
typedef __attribute__((ext_vector_type(8)))  _Float16 h8;
typedef __attribute__((ext_vector_type(8)))  float    v8f;
typedef __attribute__((ext_vector_type(4)))  float    f4;

#define T_DIM 2048
#define B_DIM 8
#define C_DIM 512
#define H_DIM 8
#define D_DIM 64
#define M_DIM (T_DIM * B_DIM)   // 16384

#if defined(__gfx1250__)
#define USE_ASYNC_LDS 1
#else
#define USE_ASYNC_LDS 0
#endif

#if USE_ASYNC_LDS
// ISA 15.18.3 opcode 98: copies 16B from global memory into LDS,
// tracked by ASYNCcnt. vdst VGPR holds the wave-relative LDS byte address
// (generic LDS pointer low 32 bits per the aperture rule addr[31:0]).
__device__ __forceinline__ void async_copy_b128_to_lds(void* lds, const void* gsrc) {
  unsigned lds_off = (unsigned)(unsigned long long)lds;
  unsigned long long ga = (unsigned long long)gsrc;
  asm volatile("global_load_async_to_lds_b128 %0, %1, off"
               :: "v"(lds_off), "v"(ga) : "memory");
}
__device__ __forceinline__ void wait_asynccnt0() {
#if __has_builtin(__builtin_amdgcn_s_wait_asynccnt)
  __builtin_amdgcn_s_wait_asynccnt(0);
#else
  asm volatile("s_wait_asynccnt 0x0" ::: "memory");
#endif
}
#endif

__device__ __forceinline__ v8f wmma_f16(v16h a, v16h b, v8f c) {
  return __builtin_amdgcn_wmma_f32_16x16x32_f16(
      false, a, false, b, (short)0, c, false, false);
}

// ---- Fragment loaders (per CDNA5 ISA 7.12.2 layouts) -----------------------
// A 16x32 f16: lane L: row = L%16; lanes 0-15 hold K={0..7,16..23},
// lanes 16-31 hold K={8..15,24..31}.
// B 32x16 f16: lane L: col = L%16; lanes 0-15 hold K={0..7,16..23}, lanes 16-31
// hold K={8..15,24..31}. B column n is a contiguous row of the row-major source
// whose rows are B's columns (W, K-tile, or V^T).

__device__ __forceinline__ v16h frag_a_from_f16(const _Float16* p, int ld) {
  const int lane = threadIdx.x & 31;
  const int row  = lane & 15;
  const int hi   = (lane >> 4) * 8;
  const _Float16* r0 = p + row * ld;
  h8 lo = *(const h8*)(r0 + hi);
  h8 hh = *(const h8*)(r0 + 16 + hi);
  v16h a;
#pragma unroll
  for (int i = 0; i < 8; ++i) { a[i] = lo[i]; a[8 + i] = hh[i]; }
  return a;
}

__device__ __forceinline__ v16h frag_b_from_f16rows(const _Float16* p, int ld) {
  const int lane = threadIdx.x & 31;
  const int col  = lane & 15;
  const int koff = (lane >> 4) * 8;
  const _Float16* r0 = p + col * ld + koff;
  h8 lo = *(const h8*)(r0);
  h8 hh = *(const h8*)(r0 + 16);
  v16h b;
#pragma unroll
  for (int i = 0; i < 8; ++i) { b[i] = lo[i]; b[8 + i] = hh[i]; }
  return b;
}

__device__ __forceinline__ float rowreduce_max(float v) {
  v = fmaxf(v, __shfl_xor(v, 1, 32));
  v = fmaxf(v, __shfl_xor(v, 2, 32));
  v = fmaxf(v, __shfl_xor(v, 4, 32));
  v = fmaxf(v, __shfl_xor(v, 8, 32));
  return v;
}
__device__ __forceinline__ float rowreduce_sum(float v) {
  v += __shfl_xor(v, 1, 32);
  v += __shfl_xor(v, 2, 32);
  v += __shfl_xor(v, 4, 32);
  v += __shfl_xor(v, 8, 32);
  return v;
}

// ---- Kernel 0: f32 -> f16 conversion pre-pass (memory bound, ~2us) ---------
__global__ __launch_bounds__(256) void cvt_f32_to_f16_kernel(
    const float* __restrict__ src, _Float16* __restrict__ dst, int n8) {
  const int i = blockIdx.x * 256 + threadIdx.x;
  if (i >= n8) return;
  f4 a = ((const f4*)src)[2 * i];
  f4 b = ((const f4*)src)[2 * i + 1];
  h8 o;
#pragma unroll
  for (int j = 0; j < 4; ++j) { o[j] = (_Float16)a[j]; o[4 + j] = (_Float16)b[j]; }
  ((h8*)dst)[i] = o;
}

// ---- Kernel 1: fused QKV projection  Y = X @ W^T + b ------------------------
// X16: [m=(t*B+b)][c] f16. Output: [B,H,T,D] f16 (Q pre-scaled by 1/8).
// Each wave: 32(M) x 64(N) tile -> 8 accumulators, 8 WMMA per k-step.
__global__ __launch_bounds__(256) void qkv_proj_kernel(
    const _Float16* __restrict__ Xq, const _Float16* __restrict__ Xk,
    const _Float16* __restrict__ Xv,
    const _Float16* __restrict__ Wq16, const _Float16* __restrict__ Wk16,
    const _Float16* __restrict__ Wv16,
    const float* __restrict__ bq, const float* __restrict__ bk,
    const float* __restrict__ bv,
    _Float16* __restrict__ Qh, _Float16* __restrict__ Kh,
    _Float16* __restrict__ Vh) {
  const int which = blockIdx.z;
  const _Float16* X; const _Float16* W; const float* bias; _Float16* out;
  float scale;
  if (which == 0)      { X = Xq; W = Wq16; bias = bq; out = Qh; scale = 0.125f; }
  else if (which == 1) { X = Xk; W = Wk16; bias = bk; out = Kh; scale = 1.0f; }
  else                 { X = Xv; W = Wv16; bias = bv; out = Vh; scale = 1.0f; }

  const int wave = threadIdx.x >> 5;
  const int lane = threadIdx.x & 31;
  const int m0 = blockIdx.x * 256 + wave * 32;
  const int n0 = blockIdx.y * 64;

  v8f acc[2][4] = {};
  for (int k0 = 0; k0 < C_DIM; k0 += 32) {
    v16h a0 = frag_a_from_f16(X + (size_t)m0 * C_DIM + k0, C_DIM);
    v16h a1 = frag_a_from_f16(X + (size_t)(m0 + 16) * C_DIM + k0, C_DIM);
#pragma unroll
    for (int j = 0; j < 4; ++j) {
      v16h b = frag_b_from_f16rows(W + (size_t)(n0 + j * 16) * C_DIM + k0, C_DIM);
      acc[0][j] = wmma_f16(a0, b, acc[0][j]);
      acc[1][j] = wmma_f16(a1, b, acc[1][j]);
    }
  }

  const int col = lane & 15;
  const int rhi = (lane >> 4) * 8;
#pragma unroll
  for (int j = 0; j < 4; ++j) {
    const int n = n0 + j * 16 + col;
    const float bb = bias[n];
    const int h = n >> 6, d = n & 63;
#pragma unroll
    for (int mi = 0; mi < 2; ++mi) {
#pragma unroll
      for (int r = 0; r < 8; ++r) {
        const int m = m0 + mi * 16 + r + rhi;
        const int t = m >> 3;          // m / B
        const int b_ = m & 7;          // m % B
        const float val = (acc[mi][j][r] + bb) * scale;
        out[(((size_t)(b_ * H_DIM + h)) * T_DIM + t) * D_DIM + d] = (_Float16)val;
      }
    }
  }
}

// ---- Kernel 2: flash attention per (b,h) ------------------------------------
__global__ __launch_bounds__(256) void attn_kernel(
    const _Float16* __restrict__ Qh, const _Float16* __restrict__ Kh,
    const _Float16* __restrict__ Vh, _Float16* __restrict__ Y) {
  __shared__ __align__(16) _Float16 Kt[32 * 64];     // keys row-major [k][d]
  __shared__ __align__(16) _Float16 VtT[64 * 32];    // V transposed [d][k]
  __shared__ __align__(16) _Float16 Pb[8][16 * 32];  // per-wave P staging

  const int bh   = blockIdx.y;                 // 0..63  (b*H + h)
  const int wave = threadIdx.x >> 5;
  const int lane = threadIdx.x & 31;
  const int tid  = threadIdx.x;
  const int q0   = blockIdx.x * 128 + wave * 16;

  const _Float16* Qb = Qh + (size_t)bh * T_DIM * D_DIM;
  const _Float16* Kb = Kh + (size_t)bh * T_DIM * D_DIM;
  const _Float16* Vb = Vh + (size_t)bh * T_DIM * D_DIM;

  v16h qa0 = frag_a_from_f16(Qb + q0 * D_DIM + 0, D_DIM);
  v16h qa1 = frag_a_from_f16(Qb + q0 * D_DIM + 32, D_DIM);

  float mrow[8], lrow[8];
  v8f O[4] = {};
#pragma unroll
  for (int r = 0; r < 8; ++r) { mrow[r] = -1e30f; lrow[r] = 0.0f; }

  for (int kb = 0; kb < T_DIM; kb += 32) {
    __syncthreads();
    {   // stage 32 keys x 64 dims; K via async VGPR-free copy, V transposed
#if USE_ASYNC_LDS
      async_copy_b128_to_lds(Kt + tid * 8, Kb + (size_t)kb * D_DIM + tid * 8);
#else
      ((h8*)Kt)[tid] = ((const h8*)(Kb + kb * D_DIM))[tid];
#endif
      h8 v = ((const h8*)(Vb + kb * D_DIM))[tid];
      const int krow = (tid * 8) >> 6;
      const int d0   = (tid * 8) & 63;
#pragma unroll
      for (int j = 0; j < 8; ++j) VtT[(d0 + j) * 32 + krow] = v[j];
      if (kb + 32 < T_DIM) {   // gfx1250 global_prefetch of next tile
        __builtin_prefetch(Kb + (kb + 32) * D_DIM + tid * 8, 0, 1);
        __builtin_prefetch(Vb + (kb + 32) * D_DIM + tid * 8, 0, 1);
      }
#if USE_ASYNC_LDS
      wait_asynccnt0();
#endif
    }
    __syncthreads();

    v8f s0 = {}, s1 = {};
    {
      v16h b00 = frag_b_from_f16rows(Kt + 0 * 64 + 0, 64);
      v16h b01 = frag_b_from_f16rows(Kt + 0 * 64 + 32, 64);
      s0 = wmma_f16(qa0, b00, s0);
      s0 = wmma_f16(qa1, b01, s0);
      v16h b10 = frag_b_from_f16rows(Kt + 16 * 64 + 0, 64);
      v16h b11 = frag_b_from_f16rows(Kt + 16 * 64 + 32, 64);
      s1 = wmma_f16(qa0, b10, s1);
      s1 = wmma_f16(qa1, b11, s1);
    }

    _Float16* pw = Pb[wave];
    const int col = lane & 15;
    const int rhi = (lane >> 4) * 8;
#pragma unroll
    for (int r = 0; r < 8; ++r) {
      float tmax = rowreduce_max(fmaxf(s0[r], s1[r]));
      float nm = fmaxf(mrow[r], tmax);
      float alpha = __expf(mrow[r] - nm);
      float p0 = __expf(s0[r] - nm);
      float p1 = __expf(s1[r] - nm);
      float rs = rowreduce_sum(p0 + p1);
      lrow[r] = lrow[r] * alpha + rs;
      mrow[r] = nm;
#pragma unroll
      for (int dt = 0; dt < 4; ++dt) O[dt][r] *= alpha;
      const int prow = r + rhi;
      pw[prow * 32 + col]      = (_Float16)p0;
      pw[prow * 32 + 16 + col] = (_Float16)p1;
    }
    __syncthreads();

    v16h pa = frag_a_from_f16(pw, 32);
#pragma unroll
    for (int dt = 0; dt < 4; ++dt) {
      v16h vb = frag_b_from_f16rows(VtT + (dt * 16) * 32, 32);
      O[dt] = wmma_f16(pa, vb, O[dt]);
    }
  }

  const int b_ = bh >> 3, h = bh & 7;
  const int col = lane & 15;
  const int rhi = (lane >> 4) * 8;
#pragma unroll
  for (int dt = 0; dt < 4; ++dt) {
#pragma unroll
    for (int r = 0; r < 8; ++r) {
      const int t = q0 + r + rhi;
      const int d = dt * 16 + col;
      const float val = O[dt][r] / lrow[r];
      Y[((size_t)(t * B_DIM + b_)) * C_DIM + h * D_DIM + d] = (_Float16)val;
    }
  }
}

// ---- Kernel 3: output projection  out = Y @ Wo^T + bo,  f32 [T,B,C] ---------
__global__ __launch_bounds__(256) void out_proj_kernel(
    const _Float16* __restrict__ Y, const _Float16* __restrict__ Wo16,
    const float* __restrict__ bo, float* __restrict__ out) {
  const int wave = threadIdx.x >> 5;
  const int lane = threadIdx.x & 31;
  const int m0 = blockIdx.x * 256 + wave * 32;
  const int n0 = blockIdx.y * 64;

  v8f acc[2][4] = {};
  for (int k0 = 0; k0 < C_DIM; k0 += 32) {
    v16h a0 = frag_a_from_f16(Y + (size_t)m0 * C_DIM + k0, C_DIM);
    v16h a1 = frag_a_from_f16(Y + (size_t)(m0 + 16) * C_DIM + k0, C_DIM);
#pragma unroll
    for (int j = 0; j < 4; ++j) {
      v16h b = frag_b_from_f16rows(Wo16 + (size_t)(n0 + j * 16) * C_DIM + k0, C_DIM);
      acc[0][j] = wmma_f16(a0, b, acc[0][j]);
      acc[1][j] = wmma_f16(a1, b, acc[1][j]);
    }
  }

  const int col = lane & 15;
  const int rhi = (lane >> 4) * 8;
#pragma unroll
  for (int j = 0; j < 4; ++j) {
    const int n = n0 + j * 16 + col;
    const float bb = bo[n];
#pragma unroll
    for (int mi = 0; mi < 2; ++mi) {
#pragma unroll
      for (int r = 0; r < 8; ++r) {
        const int m = m0 + mi * 16 + r + rhi;
        out[(size_t)m * C_DIM + n] = acc[mi][j][r] + bb;
      }
    }
  }
}

// ---- launch -----------------------------------------------------------------
extern "C" void kernel_launch(void* const* d_in, const int* in_sizes, int n_in,
                              void* d_out, int out_size, void* d_ws, size_t ws_size,
                              hipStream_t stream) {
  const float* query = (const float*)d_in[0];
  const float* key   = (const float*)d_in[1];
  const float* value = (const float*)d_in[2];
  const float* Wq = (const float*)d_in[3];
  const float* bq = (const float*)d_in[4];
  const float* Wk = (const float*)d_in[5];
  const float* bk = (const float*)d_in[6];
  const float* Wv = (const float*)d_in[7];
  const float* bv = (const float*)d_in[8];
  const float* Wo = (const float*)d_in[9];
  const float* bo = (const float*)d_in[10];
  float* out = (float*)d_out;

  const size_t S = (size_t)M_DIM * C_DIM;     // 8,388,608 activations
  const size_t W = (size_t)C_DIM * C_DIM;     // 262,144 weights

  _Float16* Xq16 = (_Float16*)d_ws;           // later reused as Y
  _Float16* Xk16 = Xq16 + S;
  _Float16* Xv16 = Xq16 + 2 * S;
  _Float16* Wq16 = Xq16 + 3 * S;
  _Float16* Wk16 = Wq16 + W;
  _Float16* Wv16 = Wq16 + 2 * W;
  _Float16* Wo16 = Wq16 + 3 * W;
  _Float16* Qh   = Wq16 + 4 * W;
  _Float16* Kh   = Qh + S;
  _Float16* Vh   = Qh + 2 * S;
  _Float16* Y    = Xq16;                      // alias: Xq16 dead after QKV

  const int actBlocks = (int)((S / 8 + 255) / 256);
  const int wBlocks   = (int)((W / 8 + 255) / 256);
  cvt_f32_to_f16_kernel<<<actBlocks, 256, 0, stream>>>(query, Xq16, (int)(S / 8));
  cvt_f32_to_f16_kernel<<<actBlocks, 256, 0, stream>>>(key,   Xk16, (int)(S / 8));
  cvt_f32_to_f16_kernel<<<actBlocks, 256, 0, stream>>>(value, Xv16, (int)(S / 8));
  cvt_f32_to_f16_kernel<<<wBlocks, 256, 0, stream>>>(Wq, Wq16, (int)(W / 8));
  cvt_f32_to_f16_kernel<<<wBlocks, 256, 0, stream>>>(Wk, Wk16, (int)(W / 8));
  cvt_f32_to_f16_kernel<<<wBlocks, 256, 0, stream>>>(Wv, Wv16, (int)(W / 8));
  cvt_f32_to_f16_kernel<<<wBlocks, 256, 0, stream>>>(Wo, Wo16, (int)(W / 8));

  qkv_proj_kernel<<<dim3(M_DIM / 256, C_DIM / 64, 3), 256, 0, stream>>>(
      Xq16, Xk16, Xv16, Wq16, Wk16, Wv16, bq, bk, bv, Qh, Kh, Vh);
  attn_kernel<<<dim3(T_DIM / 128, B_DIM * H_DIM), 256, 0, stream>>>(Qh, Kh, Vh, Y);
  out_proj_kernel<<<dim3(M_DIM / 256, C_DIM / 64), 256, 0, stream>>>(Y, Wo16, bo, out);
}
